// RWKV6Layer_31516470018819
// MI455X (gfx1250) — compile-verified, matrix operand order checked by reading, not run on hardware
//
#include <hip/hip_runtime.h>
#include <cstdint>
#include <cstddef>

// ---------------------------------------------------------------------------
// Types for CDNA5 WMMA (gfx1250, wave32)
// ---------------------------------------------------------------------------
typedef __attribute__((ext_vector_type(16))) __bf16 v16bf;
typedef __attribute__((ext_vector_type(8)))  __bf16 v8bf;
typedef __attribute__((ext_vector_type(8)))  float  v8f;

union V16 { v16bf v; v8bf h[2]; };

// Explicit global-address-space pointer: forces global_load_b128 (not flat).
typedef __attribute__((address_space(1))) const v8bf gv8bf;

// Model dims (fixed by the reference)
#define Bv   8
#define Tv   1024
#define Hv   512
#define DKv  256
#define DVv  512
#define NHv  4
#define HKv  64
#define HVv  128
#define RLv  32
#define WLv  64
#define IDv  1024
#define BTv  (Bv * Tv)

// ---------------------------------------------------------------------------
// Generic bf16 WMMA GEMM:  C[M,N](f32) = A[M,lda](bf16) * W[N,ldb](bf16)^T
// grid.x = M/128, grid.y = ceil(N/64), block = 128 threads (4 waves).
// Each wave: 32 rows x 64 cols via 8 f32 accumulators. Branch-free inner
// loop: all 6 fragments loaded up front (one clause, one wait), then 8
// back-to-back WMMAs; load columns clamped to N-1, stores guarded.
// ---------------------------------------------------------------------------
__global__ __launch_bounds__(128)
void gemm_bf16_nt(const __bf16* __restrict__ A, int lda,
                  const __bf16* __restrict__ W, int ldb,
                  float* __restrict__ C, int N, int K) {
  const int wave = threadIdx.x >> 5;
  const int lane = threadIdx.x & 31;
  const int half = lane >> 4;       // which 16-lane half (K sub-pattern)
  const int l16  = lane & 15;
  const int rowBase = blockIdx.x * 128 + wave * 32;
  const int colBase = blockIdx.y * 64;

  const int r0 = rowBase + l16;
  const int r1 = r0 + 16;

  // Per-fragment base pointers in the GLOBAL address space; columns clamped
  // so loads are always in-bounds (stores are guarded below).
  gv8bf* pa0 = (gv8bf*)(A + (size_t)r0 * lda + 8 * half);
  gv8bf* pa1 = (gv8bf*)(A + (size_t)r1 * lda + 8 * half);
  int c0 = colBase + l16;           c0 = (c0 < N) ? c0 : (N - 1);
  int c1 = colBase + 16 + l16;      c1 = (c1 < N) ? c1 : (N - 1);
  int c2 = colBase + 32 + l16;      c2 = (c2 < N) ? c2 : (N - 1);
  int c3 = colBase + 48 + l16;      c3 = (c3 < N) ? c3 : (N - 1);
  gv8bf* pb0 = (gv8bf*)(W + (size_t)c0 * ldb + 8 * half);
  gv8bf* pb1 = (gv8bf*)(W + (size_t)c1 * ldb + 8 * half);
  gv8bf* pb2 = (gv8bf*)(W + (size_t)c2 * ldb + 8 * half);
  gv8bf* pb3 = (gv8bf*)(W + (size_t)c3 * ldb + 8 * half);

  v8f acc[8] = {};
  for (int kb = 0; kb < K; kb += 32) {
    V16 a0, a1, b0, b1, b2, b3;
    a0.h[0] = pa0[0]; a0.h[1] = pa0[2];   // K = k0..k0+7, k0+16..k0+23
    a1.h[0] = pa1[0]; a1.h[1] = pa1[2];
    b0.h[0] = pb0[0]; b0.h[1] = pb0[2];
    b1.h[0] = pb1[0]; b1.h[1] = pb1[2];
    b2.h[0] = pb2[0]; b2.h[1] = pb2[2];
    b3.h[0] = pb3[0]; b3.h[1] = pb3[2];
    __builtin_prefetch((const __bf16*)(pa0 + 8), 0, 3);  // next 64 K of row

    acc[0] = __builtin_amdgcn_wmma_f32_16x16x32_bf16(false, a0.v, false, b0.v,
                                                     (short)0, acc[0], false, false);
    acc[4] = __builtin_amdgcn_wmma_f32_16x16x32_bf16(false, a1.v, false, b0.v,
                                                     (short)0, acc[4], false, false);
    acc[1] = __builtin_amdgcn_wmma_f32_16x16x32_bf16(false, a0.v, false, b1.v,
                                                     (short)0, acc[1], false, false);
    acc[5] = __builtin_amdgcn_wmma_f32_16x16x32_bf16(false, a1.v, false, b1.v,
                                                     (short)0, acc[5], false, false);
    acc[2] = __builtin_amdgcn_wmma_f32_16x16x32_bf16(false, a0.v, false, b2.v,
                                                     (short)0, acc[2], false, false);
    acc[6] = __builtin_amdgcn_wmma_f32_16x16x32_bf16(false, a1.v, false, b2.v,
                                                     (short)0, acc[6], false, false);
    acc[3] = __builtin_amdgcn_wmma_f32_16x16x32_bf16(false, a0.v, false, b3.v,
                                                     (short)0, acc[3], false, false);
    acc[7] = __builtin_amdgcn_wmma_f32_16x16x32_bf16(false, a1.v, false, b3.v,
                                                     (short)0, acc[7], false, false);
    pa0 += 4; pa1 += 4;                   // advance 32 bf16
    pb0 += 4; pb1 += 4; pb2 += 4; pb3 += 4;
  }

#pragma unroll
  for (int nt = 0; nt < 4; nt++) {
    const int col = colBase + nt * 16 + l16;
    if (col < N) {
#pragma unroll
      for (int i = 0; i < 8; i++) {
        const int row = rowBase + i + 8 * half;   // C layout: VGPR i -> M=i(+8)
        C[(size_t)row * N + col] = acc[nt][i];
        C[(size_t)(row + 16) * N + col] = acc[4 + nt][i];
      }
    }
  }
}

// ---------------------------------------------------------------------------
// Elementwise / small kernels
// ---------------------------------------------------------------------------
__global__ void cvt_bf16_k(const float* __restrict__ s, __bf16* __restrict__ d, int n) {
  int i = blockIdx.x * 256 + threadIdx.x;
  if (i < n) d[i] = (__bf16)s[i];
}

// LayerNorm over last dim (width Hd), one block per row, 256 threads.
__global__ __launch_bounds__(256)
void ln_k(const float* __restrict__ x, const float* __restrict__ g,
          const float* __restrict__ b, float* __restrict__ y, int Hd) {
  const int row = blockIdx.x, tid = threadIdx.x;
  const float* xr = x + (size_t)row * Hd;
  float s = 0.f, s2 = 0.f;
  for (int i = tid; i < Hd; i += 256) { float v = xr[i]; s += v; s2 += v * v; }
  for (int off = 16; off; off >>= 1) { s += __shfl_xor(s, off); s2 += __shfl_xor(s2, off); }
  __shared__ float rs[8], rq[8];
  if ((tid & 31) == 0) { rs[tid >> 5] = s; rq[tid >> 5] = s2; }
  __syncthreads();
  float S = 0.f, S2 = 0.f;
#pragma unroll
  for (int i = 0; i < 8; i++) { S += rs[i]; S2 += rq[i]; }
  const float mean = S / Hd;
  const float inv = rsqrtf(S2 / Hd - mean * mean + 1e-5f);
  float* yr = y + (size_t)row * Hd;
  for (int i = tid; i < Hd; i += 256) yr[i] = (xr[i] - mean) * inv * g[i] + b[i];
}

// token-shift mix: out = h + (h[t-1]-h)*mu   (bf16 out)
__global__ void tsmix_k(const float* __restrict__ h, const float* __restrict__ mu,
                        __bf16* __restrict__ out, int n, int Hd, int T) {
  int i = blockIdx.x * 256 + threadIdx.x;
  if (i >= n) return;
  const int c = i % Hd;
  const int t = (i / Hd) % T;
  const float hv = h[i];
  const float hp = (t > 0) ? h[i - Hd] : 0.f;
  out[i] = (__bf16)(hv + (hp - hv) * mu[c]);
}

// data-dependent mix: out = h + (h[t-1]-h)*(zraw + bias)   (bf16 out)
__global__ void zmix_k(const float* __restrict__ h, const float* __restrict__ zraw,
                       const float* __restrict__ bias, __bf16* __restrict__ out,
                       int n, int Hd, int T) {
  int i = blockIdx.x * 256 + threadIdx.x;
  if (i >= n) return;
  const int c = i % Hd;
  const int t = (i / Hd) % T;
  const float hv = h[i];
  const float hp = (t > 0) ? h[i - Hd] : 0.f;
  out[i] = (__bf16)(hv + (hp - hv) * (zraw[i] + bias[c]));
}

__global__ void tanh_bf16_k(const float* __restrict__ s, __bf16* __restrict__ d, int n) {
  int i = blockIdx.x * 256 + threadIdx.x;
  if (i < n) d[i] = (__bf16)tanhf(s[i]);
}

// lw = -exp(wraw + w_b)  (in place, width DK)
__global__ void lw_k(float* __restrict__ w, const float* __restrict__ wb, int n) {
  int i = blockIdx.x * 256 + threadIdx.x;
  if (i < n) w[i] = -expf(w[i] + wb[i % DKv]);
}

__global__ void relu2_bf16_k(const float* __restrict__ s, __bf16* __restrict__ d, int n) {
  int i = blockIdx.x * 256 + threadIdx.x;
  if (i < n) { float r = s[i]; r = r > 0.f ? r : 0.f; d[i] = (__bf16)(r * r); }
}

__global__ void resadd_k(const float* __restrict__ x0, const float* __restrict__ a,
                         float* __restrict__ res, int n) {
  int i = blockIdx.x * 256 + threadIdx.x;
  if (i < n) res[i] = x0[i] + a[i];
}

__global__ void final_k(const float* __restrict__ res, const float* __restrict__ rec,
                        const float* __restrict__ val, float* __restrict__ out, int n) {
  int i = blockIdx.x * 256 + threadIdx.x;
  if (i < n) out[i] = res[i] + val[i] / (1.f + expf(-rec[i]));
}

// ---------------------------------------------------------------------------
// WKV6 selective-decay recurrence.  State S[k][v] independent per (k,v);
// o[v] = sum_k r[k]*(S + bonus*k*v).  Parallel over B*NH*(HV/32) blocks;
// each thread holds 8 state elements, 8-lane shuffle reduction over k.
// ---------------------------------------------------------------------------
__global__ __launch_bounds__(256)
void wkv6_scan_k(const float* __restrict__ r, const float* __restrict__ k,
                 const float* __restrict__ v, const float* __restrict__ lw,
                 const float* __restrict__ bonus, float* __restrict__ o) {
  const int blk  = blockIdx.x;
  const int vblk = blk & 3;          // HV/32 slices
  const int head = (blk >> 2) & 3;   // NH
  const int b    = blk >> 4;
  const int tid  = threadIdx.x;
  const int c    = tid >> 3;         // 0..31 : v column within slice
  const int kg   = tid & 7;          // 8 threads cover 64 k each

  __shared__ float r_s[HKv], k_s[HKv], ew_s[HKv], v_s[32], bon_s[HKv];
  if (tid < HKv) bon_s[tid] = bonus[head * HKv + tid];
  __syncthreads();

  float S[8];
#pragma unroll
  for (int j = 0; j < 8; j++) S[j] = 0.f;

  size_t rkBase = ((size_t)b * Tv) * NHv * HKv + (size_t)head * HKv;
  size_t vBase  = ((size_t)b * Tv) * NHv * HVv + (size_t)head * HVv + vblk * 32;

  for (int t = 0; t < Tv; t++) {
    if (tid < HKv) {
      r_s[tid]  = r[rkBase + tid];
      k_s[tid]  = k[rkBase + tid];
      ew_s[tid] = expf(lw[rkBase + tid]);
    } else if (tid < HKv + 32) {
      v_s[tid - HKv] = v[vBase + (tid - HKv)];
    }
    __syncthreads();
    const float vv = v_s[c];
    float part = 0.f;
#pragma unroll
    for (int j = 0; j < 8; j++) {
      const int ki = kg * 8 + j;
      const float kv = k_s[ki] * vv;
      part += r_s[ki] * (S[j] + bon_s[ki] * kv);
      S[j] = ew_s[ki] * S[j] + kv;
    }
    part += __shfl_xor(part, 1);
    part += __shfl_xor(part, 2);
    part += __shfl_xor(part, 4);
    if (kg == 0) o[vBase + c] = part;
    __syncthreads();
    rkBase += NHv * HKv;
    vBase  += NHv * HVv;
  }
}

// GroupNorm over HV per (b,t,head) + swish gate -> bf16 for o_W GEMM.
__global__ __launch_bounds__(128)
void gnorm_gate_k(const float* __restrict__ o, const float* __restrict__ g,
                  const float* __restrict__ gg, const float* __restrict__ gb,
                  __bf16* __restrict__ gated) {
  const int grp = blockIdx.x;        // bt*NH + head
  const int head = grp & 3;
  const size_t bt = grp >> 2;
  const int tid = threadIdx.x;       // v index 0..127
  const size_t idx = bt * DVv + (size_t)head * HVv + tid;
  const float x = o[idx];
  float s = x, s2 = x * x;
  for (int off = 16; off; off >>= 1) { s += __shfl_xor(s, off); s2 += __shfl_xor(s2, off); }
  __shared__ float rs[4], rq[4];
  if ((tid & 31) == 0) { rs[tid >> 5] = s; rq[tid >> 5] = s2; }
  __syncthreads();
  const float S  = rs[0] + rs[1] + rs[2] + rs[3];
  const float S2 = rq[0] + rq[1] + rq[2] + rq[3];
  const float mean = S * (1.f / 128.f);
  const float inv = rsqrtf(S2 * (1.f / 128.f) - mean * mean + 1e-5f);
  const int ch = head * HVv + tid;
  const float gv = g[idx];
  const float sw = gv / (1.f + expf(-gv));
  gated[idx] = (__bf16)(((x - mean) * inv * gg[ch] + gb[ch]) * sw);
}

// ---------------------------------------------------------------------------
// Host launcher
// ---------------------------------------------------------------------------
extern "C" void kernel_launch(void* const* d_in, const int* in_sizes, int n_in,
                              void* d_out, int out_size, void* d_ws, size_t ws_size,
                              hipStream_t stream) {
  const float* x        = (const float*)d_in[0];
  const float* pre_g    = (const float*)d_in[1];
  const float* pre_b    = (const float*)d_in[2];
  const float* attn_g   = (const float*)d_in[3];
  const float* attn_b   = (const float*)d_in[4];
  const float* ffn_g    = (const float*)d_in[5];
  const float* ffn_b    = (const float*)d_in[6];
  const float* xproj_mu = (const float*)d_in[7];
  const float* xproj_w1 = (const float*)d_in[8];
  const float* xproj_w2 = (const float*)d_in[9];
  const float* x_bias   = (const float*)d_in[10];
  const float* r_W      = (const float*)d_in[11];
  const float* w_W1     = (const float*)d_in[12];
  const float* w_W2     = (const float*)d_in[13];
  const float* w_b      = (const float*)d_in[14];
  const float* k_W      = (const float*)d_in[15];
  const float* v_W      = (const float*)d_in[16];
  const float* g_W      = (const float*)d_in[17];
  const float* bonus    = (const float*)d_in[18];
  const float* gnorm_g  = (const float*)d_in[19];
  const float* gnorm_b  = (const float*)d_in[20];
  const float* o_W      = (const float*)d_in[21];
  const float* fk_mu    = (const float*)d_in[22];
  const float* fk_W     = (const float*)d_in[23];
  const float* fv_W     = (const float*)d_in[24];
  const float* fr_mu    = (const float*)d_in[25];
  const float* fr_W     = (const float*)d_in[26];
  float* out = (float*)d_out;

  // bump allocator over d_ws
  char* ws = (char*)d_ws;
  size_t off = 0;
  auto alloc = [&](size_t bytes) -> void* {
    off = (off + 255) & ~(size_t)255;
    void* p = ws + off;
    off += bytes;
    return p;
  };
  const size_t F = sizeof(float), BF = 2;

  // f32 intermediates
  float* x0    = (float*)alloc((size_t)BTv * Hv * F);
  float* h     = (float*)alloc((size_t)BTv * Hv * F);
  float* z1raw = (float*)alloc((size_t)BTv * 5 * RLv * F);
  float* ztmp  = (float*)alloc((size_t)BTv * Hv * F);
  float* rbuf  = (float*)alloc((size_t)BTv * DKv * F);
  float* wtraw = (float*)alloc((size_t)BTv * WLv * F);
  float* wraw  = (float*)alloc((size_t)BTv * DKv * F);   // becomes lw in place
  float* kbuf  = (float*)alloc((size_t)BTv * DKv * F);
  float* vbuf  = (float*)alloc((size_t)BTv * DVv * F);
  float* gbuf  = (float*)alloc((size_t)BTv * DVv * F);
  float* obuf  = (float*)alloc((size_t)BTv * DVv * F);
  float* abuf  = (float*)alloc((size_t)BTv * Hv * F);
  float* res   = (float*)alloc((size_t)BTv * Hv * F);
  float* hf    = (float*)alloc((size_t)BTv * Hv * F);
  float* kkraw = (float*)alloc((size_t)BTv * IDv * F);
  float* val   = (float*)alloc((size_t)BTv * Hv * F);
  float* recr  = (float*)alloc((size_t)BTv * Hv * F);

  // bf16 activations
  __bf16* mix0  = (__bf16*)alloc((size_t)BTv * Hv * BF);
  __bf16* z1bf  = (__bf16*)alloc((size_t)BTv * 5 * RLv * BF);
  __bf16* mixr  = (__bf16*)alloc((size_t)BTv * Hv * BF);
  __bf16* mixw  = (__bf16*)alloc((size_t)BTv * Hv * BF);
  __bf16* mixk  = (__bf16*)alloc((size_t)BTv * Hv * BF);
  __bf16* mixv  = (__bf16*)alloc((size_t)BTv * Hv * BF);
  __bf16* mixg  = (__bf16*)alloc((size_t)BTv * Hv * BF);
  __bf16* wtbf  = (__bf16*)alloc((size_t)BTv * WLv * BF);
  __bf16* gated = (__bf16*)alloc((size_t)BTv * DVv * BF);
  __bf16* fkx   = (__bf16*)alloc((size_t)BTv * Hv * BF);
  __bf16* frx   = (__bf16*)alloc((size_t)BTv * Hv * BF);
  __bf16* kkbf  = (__bf16*)alloc((size_t)BTv * IDv * BF);

  // bf16 weights
  __bf16* xw1b = (__bf16*)alloc((size_t)5 * RLv * Hv * BF);
  __bf16* xw2b = (__bf16*)alloc((size_t)Hv * 5 * RLv * BF);
  __bf16* rWb  = (__bf16*)alloc((size_t)DKv * Hv * BF);
  __bf16* wW1b = (__bf16*)alloc((size_t)WLv * Hv * BF);
  __bf16* wW2b = (__bf16*)alloc((size_t)DKv * WLv * BF);
  __bf16* kWb  = (__bf16*)alloc((size_t)DKv * Hv * BF);
  __bf16* vWb  = (__bf16*)alloc((size_t)DVv * Hv * BF);
  __bf16* gWb  = (__bf16*)alloc((size_t)DVv * Hv * BF);
  __bf16* oWb  = (__bf16*)alloc((size_t)Hv * DVv * BF);
  __bf16* fkWb = (__bf16*)alloc((size_t)IDv * Hv * BF);
  __bf16* fvWb = (__bf16*)alloc((size_t)Hv * IDv * BF);
  __bf16* frWb = (__bf16*)alloc((size_t)Hv * Hv * BF);

  auto cvt = [&](const float* s, __bf16* d, size_t n) {
    cvt_bf16_k<<<(int)((n + 255) / 256), 256, 0, stream>>>(s, d, (int)n);
  };
  auto gemm = [&](const __bf16* A, int lda, const __bf16* W, int ldb,
                  float* C, int N, int K) {
    dim3 g(BTv / 128, (N + 63) / 64);
    gemm_bf16_nt<<<g, 128, 0, stream>>>(A, lda, W, ldb, C, N, K);
  };

  // weights -> bf16
  cvt(xproj_w1, xw1b, (size_t)5 * RLv * Hv);
  cvt(xproj_w2, xw2b, (size_t)Hv * 5 * RLv);
  cvt(r_W,  rWb,  (size_t)DKv * Hv);
  cvt(w_W1, wW1b, (size_t)WLv * Hv);
  cvt(w_W2, wW2b, (size_t)DKv * WLv);
  cvt(k_W,  kWb,  (size_t)DKv * Hv);
  cvt(v_W,  vWb,  (size_t)DVv * Hv);
  cvt(g_W,  gWb,  (size_t)DVv * Hv);
  cvt(o_W,  oWb,  (size_t)Hv * DVv);
  cvt(fk_W, fkWb, (size_t)IDv * Hv);
  cvt(fv_W, fvWb, (size_t)Hv * IDv);
  cvt(fr_W, frWb, (size_t)Hv * Hv);

  const int nBH = BTv * Hv;
  const int gBH = (nBH + 255) / 256;

  // pre-norm + attn-norm
  ln_k<<<BTv, 256, 0, stream>>>(x,  pre_g,  pre_b,  x0, Hv);
  ln_k<<<BTv, 256, 0, stream>>>(x0, attn_g, attn_b, h,  Hv);

  // low-rank interpolation factors
  tsmix_k<<<gBH, 256, 0, stream>>>(h, xproj_mu, mix0, nBH, Hv, Tv);
  gemm(mix0, Hv, xw1b, Hv, z1raw, 5 * RLv, Hv);                // [BT,160]
  tanh_bf16_k<<<(BTv * 5 * RLv + 255) / 256, 256, 0, stream>>>(z1raw, z1bf,
                                                               BTv * 5 * RLv);
  __bf16* mixes[5] = {mixr, mixw, mixk, mixv, mixg};
  for (int n5 = 0; n5 < 5; n5++) {
    // z_n = z1[:, n, :] @ xproj_w2[:, n, :]^T  (K = 32)
    gemm(z1bf + n5 * RLv, 5 * RLv, xw2b + n5 * RLv, 5 * RLv, ztmp, Hv, RLv);
    zmix_k<<<gBH, 256, 0, stream>>>(h, ztmp, x_bias + n5 * Hv, mixes[n5],
                                    nBH, Hv, Tv);
  }

  // projections
  gemm(mixr, Hv, rWb, Hv, rbuf, DKv, Hv);
  gemm(mixw, Hv, wW1b, Hv, wtraw, WLv, Hv);
  tanh_bf16_k<<<(BTv * WLv + 255) / 256, 256, 0, stream>>>(wtraw, wtbf, BTv * WLv);
  gemm(wtbf, WLv, wW2b, WLv, wraw, DKv, WLv);
  lw_k<<<(BTv * DKv + 255) / 256, 256, 0, stream>>>(wraw, w_b, BTv * DKv);
  gemm(mixk, Hv, kWb, Hv, kbuf, DKv, Hv);
  gemm(mixv, Hv, vWb, Hv, vbuf, DVv, Hv);
  gemm(mixg, Hv, gWb, Hv, gbuf, DVv, Hv);

  // WKV6 recurrence
  wkv6_scan_k<<<Bv * NHv * (HVv / 32), 256, 0, stream>>>(rbuf, kbuf, vbuf,
                                                         wraw, bonus, obuf);

  // GroupNorm + swish gate, output projection, residual
  gnorm_gate_k<<<BTv * NHv, 128, 0, stream>>>(obuf, gbuf, gnorm_g, gnorm_b, gated);
  gemm(gated, DVv, oWb, DVv, abuf, Hv, DVv);
  resadd_k<<<gBH, 256, 0, stream>>>(x0, abuf, res, nBH);

  // channel-mix FFN
  ln_k<<<BTv, 256, 0, stream>>>(res, ffn_g, ffn_b, hf, Hv);
  tsmix_k<<<gBH, 256, 0, stream>>>(hf, fk_mu, fkx, nBH, Hv, Tv);
  tsmix_k<<<gBH, 256, 0, stream>>>(hf, fr_mu, frx, nBH, Hv, Tv);
  gemm(fkx, Hv, fkWb, Hv, kkraw, IDv, Hv);
  relu2_bf16_k<<<(BTv * IDv + 255) / 256, 256, 0, stream>>>(kkraw, kkbf, BTv * IDv);
  gemm(kkbf, IDv, fvWb, IDv, val, Hv, IDv);
  gemm(frx, Hv, frWb, Hv, recr, Hv, Hv);
  final_k<<<gBH, 256, 0, stream>>>(res, recr, val, out, nBH);
}
